// Attention_47665547051448
// MI455X (gfx1250) — compile-verified, matrix operand order checked by reading, not run on hardware
//
#include <hip/hip_runtime.h>
#include <hip/hip_fp16.h>

// ---------------------------------------------------------------------------
// Transformer attention block (B=8, N=1024, D=1024, H=16, DH=64) for gfx1250.
// ~103 GFLOP vs ~150MB HBM traffic (AI ~ 700 FLOP/B) -> compute bound.
// - all matmuls on v_wmma_f32_16x16x32_f16
// - K/V^T tiles double-buffered via global_load_async_to_lds_b128 (ASYNCcnt)
// - attention transposed (S^T = K@Q^T, O^T = V^T@P): per-lane softmax state,
//   P stays in registers as a ready-made B-fragment
// - GEMMs emit vectorized b128 stores (transposed orientation for Q/K/Wo,
//   head-transposed V layout for the PV matmul)
// ---------------------------------------------------------------------------

typedef _Float16 f16x8  __attribute__((ext_vector_type(8)));
typedef _Float16 f16x16 __attribute__((ext_vector_type(16)));
typedef float    f32x8  __attribute__((ext_vector_type(8)));

#define NB     8
#define NSEQ   1024
#define DMODEL 1024
#define NHEAD  16
#define DHEAD  64
#define EPSV   1e-5f

#define K_LD   72   // LDS row stride (halves), K tile 32x64: 144B rows
#define VT_LD  40   // LDS row stride (halves), V^T tile 64x32: 80B rows

#define WMMA16(a, b, c) \
  __builtin_amdgcn_wmma_f32_16x16x32_f16(false, (a), false, (b), (short)0, (c), false, false)

// ---------------------------------------------------------------------------
// 16x32 fp16 WMMA A-fragment (== B^T fragment) from a row-major matrix.
// ISA layout (CDNA5 ISA): lanes 0-15 hold rows M=0..15 with
// K = {k..k+7, k+16..k+23}; lanes 16-31 the same rows with K offset +8.
// ---------------------------------------------------------------------------
__device__ __forceinline__ f16x16 load_frag_g(const _Float16* base,
                                              int row0, int ld, int k0, int lane) {
  const int r  = row0 + (lane & 15);
  const int kk = k0 + ((lane >> 4) << 3);
  const _Float16* p = base + (size_t)r * ld + kk;
  union { f16x16 v; f16x8 h[2]; } u;
  u.h[0] = *(const f16x8*)(p);
  u.h[1] = *(const f16x8*)(p + 16);
  return u.v;
}

// ---------------------------------------------------------------------------
// Async global -> LDS staging (CDNA5 ASYNCcnt path, CDNA5 ISA).
// ---------------------------------------------------------------------------
__device__ __forceinline__ uint32_t lds_off(const void* p) {
  return (uint32_t)(uintptr_t)p;   // generic LDS addr: byte offset in low bits
}
__device__ __forceinline__ void async_ld_b128(uint32_t dst_lds, const void* gsrc) {
  asm volatile("global_load_async_to_lds_b128 %0, %1, off"
               :: "v"(dst_lds), "v"(gsrc) : "memory");
}
__device__ __forceinline__ void wait_async0() {
  asm volatile("s_wait_asynccnt 0" ::: "memory");
}

// ---------------------------------------------------------------------------
// f32 -> f16 conversion
// ---------------------------------------------------------------------------
__global__ void cvt_f32_f16(const float* __restrict__ s, _Float16* __restrict__ d, int n) {
  int i = blockIdx.x * blockDim.x + threadIdx.x;
  if (i < n) d[i] = (_Float16)s[i];
}

// ---------------------------------------------------------------------------
// Transposed-orientation GEMM: out[m, d] = sum_k X[m,k]*W[d,k] + bias[d]
// computed as D = W_frag(A) x X_frag(B), so C rows = d (i-contiguous per
// lane -> vectorized stores), C cols = m (fixed per lane).
// One wave per 32(d) x 64(m) tile: 8 WMMAs per K-step.
// ---------------------------------------------------------------------------
template <bool F32OUT>
__global__ void __launch_bounds__(256) gemm_wmma_t(
    const _Float16* __restrict__ X, const _Float16* __restrict__ W,
    const float* __restrict__ bias, _Float16* __restrict__ outH,
    float* __restrict__ outF, int M, int Nout, int K) {
  const int lane  = threadIdx.x & 31;
  const int gwave = blockIdx.x * (blockDim.x >> 5) + (threadIdx.x >> 5);
  const int ndt   = Nout >> 5;            // d-tiles of 32
  const int tm    = gwave / ndt;          // m-tile of 64
  const int td    = gwave - tm * ndt;
  const int mb = tm * 64, db = td * 32;
  if (mb >= M) return;

  f32x8 acc[8] = {};
  for (int k = 0; k < K; k += 32) {
    if (k + 64 < K)
      __builtin_prefetch(X + (size_t)(mb + (lane & 15)) * K + k + 64, 0, 0);
    const f16x16 w0 = load_frag_g(W, db,      K, k, lane);
    const f16x16 w1 = load_frag_g(W, db + 16, K, k, lane);
    const f16x16 x0 = load_frag_g(X, mb,      K, k, lane);
    const f16x16 x1 = load_frag_g(X, mb + 16, K, k, lane);
    const f16x16 x2 = load_frag_g(X, mb + 32, K, k, lane);
    const f16x16 x3 = load_frag_g(X, mb + 48, K, k, lane);
    acc[0] = WMMA16(w0, x0, acc[0]);
    acc[1] = WMMA16(w0, x1, acc[1]);
    acc[2] = WMMA16(w0, x2, acc[2]);
    acc[3] = WMMA16(w0, x3, acc[3]);
    acc[4] = WMMA16(w1, x0, acc[4]);
    acc[5] = WMMA16(w1, x1, acc[5]);
    acc[6] = WMMA16(w1, x2, acc[6]);
    acc[7] = WMMA16(w1, x3, acc[7]);
  }

  const int cl = lane & 15;
  const int hb = (lane >> 4) << 3;
#pragma unroll
  for (int a = 0; a < 2; ++a) {
    const int d0 = db + 16 * a + hb;       // 8 consecutive output features
    const float4 blo = *(const float4*)(bias + d0);
    const float4 bhi = *(const float4*)(bias + d0 + 4);
#pragma unroll
    for (int j = 0; j < 4; ++j) {
      const int mrow = mb + 16 * j + cl;   // fixed per lane
      const f32x8 A = acc[a * 4 + j];
      if (F32OUT) {
        float4 lo, hi;
        lo.x = A[0] + blo.x; lo.y = A[1] + blo.y; lo.z = A[2] + blo.z; lo.w = A[3] + blo.w;
        hi.x = A[4] + bhi.x; hi.y = A[5] + bhi.y; hi.z = A[6] + bhi.z; hi.w = A[7] + bhi.w;
        *(float4*)(outF + (size_t)mrow * Nout + d0)     = lo;
        *(float4*)(outF + (size_t)mrow * Nout + d0 + 4) = hi;
      } else {
        union { f16x8 v; _Float16 e[8]; } st;
        st.e[0] = (_Float16)(A[0] + blo.x); st.e[1] = (_Float16)(A[1] + blo.y);
        st.e[2] = (_Float16)(A[2] + blo.z); st.e[3] = (_Float16)(A[3] + blo.w);
        st.e[4] = (_Float16)(A[4] + bhi.x); st.e[5] = (_Float16)(A[5] + bhi.y);
        st.e[6] = (_Float16)(A[6] + bhi.z); st.e[7] = (_Float16)(A[7] + bhi.w);
        *(f16x8*)(outH + (size_t)mrow * Nout + d0) = st.v;
      }
    }
  }
}

// ---------------------------------------------------------------------------
// V projection: standard orientation, head-transposed f16 output
// vT[(b*H+h)*DH + d][n]  (n is i-contiguous -> single b128 store per frag).
// ---------------------------------------------------------------------------
__global__ void __launch_bounds__(256) gemm_wmma_vT(
    const _Float16* __restrict__ A, const _Float16* __restrict__ W,
    const float* __restrict__ bias, _Float16* __restrict__ outH,
    int M, int Nn, int K) {
  const int lane    = threadIdx.x & 31;
  const int gwave   = blockIdx.x * (blockDim.x >> 5) + (threadIdx.x >> 5);
  const int tiles_n = Nn >> 6;
  const int tm      = gwave / tiles_n;
  const int tn      = gwave - tm * tiles_n;
  const int m0 = tm * 32, n0 = tn * 64;
  if (m0 >= M) return;

  f32x8 acc[8] = {};
  for (int k = 0; k < K; k += 32) {
    if (k + 64 < K)
      __builtin_prefetch(A + (size_t)(m0 + (lane & 15)) * K + k + 64, 0, 0);
    const f16x16 a0 = load_frag_g(A, m0,      K, k, lane);
    const f16x16 a1 = load_frag_g(A, m0 + 16, K, k, lane);
    const f16x16 b0 = load_frag_g(W, n0,      K, k, lane);
    const f16x16 b1 = load_frag_g(W, n0 + 16, K, k, lane);
    const f16x16 b2 = load_frag_g(W, n0 + 32, K, k, lane);
    const f16x16 b3 = load_frag_g(W, n0 + 48, K, k, lane);
    acc[0] = WMMA16(a0, b0, acc[0]);
    acc[1] = WMMA16(a0, b1, acc[1]);
    acc[2] = WMMA16(a0, b2, acc[2]);
    acc[3] = WMMA16(a0, b3, acc[3]);
    acc[4] = WMMA16(a1, b0, acc[4]);
    acc[5] = WMMA16(a1, b1, acc[5]);
    acc[6] = WMMA16(a1, b2, acc[6]);
    acc[7] = WMMA16(a1, b3, acc[7]);
  }

  const int cl = lane & 15;
  const int hb = (lane >> 4) << 3;
#pragma unroll
  for (int half = 0; half < 2; ++half) {
    const int rb   = m0 + 16 * half + hb;  // global row = b*NSEQ + n
    const int bIdx = rb >> 10;
    const int nIdx = rb & (NSEQ - 1);      // i-contiguous
#pragma unroll
    for (int j = 0; j < 4; ++j) {
      const int col = n0 + 16 * j + cl;
      const float bb = bias[col];
      union { f16x8 v; _Float16 e[8]; } st;
#pragma unroll
      for (int i = 0; i < 8; ++i) st.e[i] = (_Float16)(acc[4 * half + j][i] + bb);
      const int hIdx = col >> 6, dd = col & 63;
      *(f16x8*)(outH + (((size_t)bIdx * NHEAD + hIdx) * DHEAD + dd) * NSEQ + nIdx) = st.v;
    }
  }
}

// ---------------------------------------------------------------------------
// Flash-attention, transposed orientation, double-buffered async staging.
// Block = 4 waves sharing one (b,h), covering 64 query rows. K tile (32x64)
// and V^T tile (64x32) for step i+1 are DMA'd while step i computes.
// ---------------------------------------------------------------------------
__global__ void __launch_bounds__(128) attn_wmma(
    const _Float16* __restrict__ qh, const _Float16* __restrict__ kh,
    const _Float16* __restrict__ vT, _Float16* __restrict__ oh) {
  __shared__ __align__(16) _Float16 Kt[2][32 * K_LD];
  __shared__ __align__(16) _Float16 Vt[2][64 * VT_LD];

  const int tid  = threadIdx.x;
  const int lane = tid & 31;
  const int wave = tid >> 5;
  const int blk  = blockIdx.x;               // [0, B*H*(N/64))
  const int mq   = blk & 15;
  const int h    = (blk >> 4) & 15;
  const int b    = blk >> 8;
  const size_t base   = ((size_t)b * NSEQ) * DMODEL + (size_t)h * DHEAD;
  const size_t vtbase = ((size_t)(b * NHEAD + h)) * DHEAD * NSEQ;
  const int q0 = mq * 64 + wave * 16;
  const int cl = lane & 15;
  const int hb = (lane >> 4) << 3;           // 0 or 8

  // B-fragments of Q^T (column = query), loaded once from global
  const f16x16 bq0 = load_frag_g(qh + base, q0, DMODEL, 0,  lane);
  const f16x16 bq1 = load_frag_g(qh + base, q0, DMODEL, 32, lane);

  float m = -3.0e38f, l = 0.f;
  f32x8 oacc[4] = {};                        // O^T: 4 d-chunks x (8 d-rows/lane)

  // stage key-tile kt into LDS buffer buf (4 async b128 per thread)
  auto stage = [&](int kt, int buf) {
    const int kb = kt * 32;
    const _Float16* kg  = kh + base + (size_t)kb * DMODEL;
    const _Float16* vTg = vT + vtbase + kb;
#pragma unroll
    for (int c = tid; c < 256; c += 128) {
      const int krow = c >> 3, kcol = (c & 7) * 8;
      async_ld_b128(lds_off(&Kt[buf][krow * K_LD + kcol]),
                    kg + (size_t)krow * DMODEL + kcol);
      const int vrow = c >> 2, vcol = (c & 3) * 8;
      async_ld_b128(lds_off(&Vt[buf][vrow * VT_LD + vcol]),
                    vTg + (size_t)vrow * NSEQ + vcol);
    }
  };

  stage(0, 0);
  for (int kt = 0; kt < NSEQ / 32; ++kt) {
    const int buf = kt & 1;
    wait_async0();       // this wave's DMA for tile kt has landed
    __syncthreads();     // ...and everyone else's; prev reads of buf^1 done
    if (kt + 1 < NSEQ / 32) stage(kt + 1, buf ^ 1);  // overlap with compute

    // --- S^T = K @ Q^T : A-frags of K straight from LDS (ds_load_b128) ---
    const f16x16 ak0 = load_frag_g(Kt[buf], 0,  K_LD, 0,  lane);
    const f16x16 ak1 = load_frag_g(Kt[buf], 0,  K_LD, 32, lane);
    const f16x16 ak2 = load_frag_g(Kt[buf], 16, K_LD, 0,  lane);
    const f16x16 ak3 = load_frag_g(Kt[buf], 16, K_LD, 32, lane);
    f32x8 st0 = {}, st1 = {};
    st0 = WMMA16(ak0, bq0, st0);
    st0 = WMMA16(ak1, bq1, st0);
    st1 = WMMA16(ak2, bq0, st1);
    st1 = WMMA16(ak3, bq1, st1);

    // --- per-lane online softmax over this lane's 16 of 32 keys ---
    float tm = -3.0e38f;
#pragma unroll
    for (int i = 0; i < 8; ++i) {
      st0[i] *= 0.125f;                      // 1/sqrt(64)
      st1[i] *= 0.125f;
      tm = fmaxf(tm, fmaxf(st0[i], st1[i]));
    }
    tm = fmaxf(tm, __shfl_xor(tm, 16));      // merge complementary key half
    const float mn = fmaxf(m, tm);
    const float co = __expf(m - mn);
    m = mn;

    union { f16x16 v; _Float16 e[16]; } pf;  // P: already a B-fragment
    float su = 0.f;
#pragma unroll
    for (int i = 0; i < 8; ++i) {
      const float p0 = __expf(st0[i] - mn);  // keys hb+i
      const float p1 = __expf(st1[i] - mn);  // keys 16+hb+i
      su += p0 + p1;
      pf.e[i]     = (_Float16)p0;
      pf.e[i + 8] = (_Float16)p1;
    }
    su += __shfl_xor(su, 16);
    l = l * co + su;
#pragma unroll
    for (int j = 0; j < 4; ++j)
#pragma unroll
      for (int i = 0; i < 8; ++i) oacc[j][i] *= co;

    // --- O^T += V^T @ P : V^T frags are plain row-major ds_load_b128 ---
#pragma unroll
    for (int j = 0; j < 4; ++j) {
      const f16x16 av = load_frag_g(Vt[buf], 16 * j, VT_LD, 0, lane);
      oacc[j] = WMMA16(av, pf.v, oacc[j]);
    }
  }

  // --- normalize (per-lane l) and store: d contiguous per frag -> b128 ---
  const float rl = 1.0f / l;
  const int orow = q0 + cl;
#pragma unroll
  for (int j = 0; j < 4; ++j) {
    union { f16x8 v; _Float16 e[8]; } ov;
#pragma unroll
    for (int i = 0; i < 8; ++i) ov.e[i] = (_Float16)(oacc[j][i] * rl);
    *(f16x8*)(oh + base + (size_t)orow * DMODEL + 16 * j + hb) = ov.v;
  }
}

// ---------------------------------------------------------------------------
// Epilogue: residual + LN -> ReLU+residual -> LN. One block per row.
// ---------------------------------------------------------------------------
__device__ __forceinline__ float block_sum(float v, float* red) {
  const int t = threadIdx.x;
  red[t] = v;
  __syncthreads();
  for (int off = 128; off > 0; off >>= 1) {
    if (t < off) red[t] += red[t + off];
    __syncthreads();
  }
  const float r = red[0];
  __syncthreads();
  return r;
}

__global__ void __launch_bounds__(256) fused_ln(
    const float* __restrict__ mha, const float* __restrict__ qres,
    const float* __restrict__ g1, const float* __restrict__ b1,
    const float* __restrict__ g2, const float* __restrict__ b2,
    float* __restrict__ out) {
  __shared__ float red[256];
  const int row = blockIdx.x;
  const int t   = threadIdx.x;
  const float* x1 = mha  + (size_t)row * DMODEL;
  const float* x2 = qres + (size_t)row * DMODEL;

  float v[4];
#pragma unroll
  for (int j = 0; j < 4; ++j) v[j] = x1[t + 256 * j] + x2[t + 256 * j];

  float s = v[0] + v[1] + v[2] + v[3];
  const float mu = block_sum(s, red) * (1.0f / DMODEL);
  float vs = 0.f;
#pragma unroll
  for (int j = 0; j < 4; ++j) { const float d = v[j] - mu; vs += d * d; }
  const float inv = rsqrtf(block_sum(vs, red) * (1.0f / DMODEL) + EPSV);

  float y[4];
#pragma unroll
  for (int j = 0; j < 4; ++j) {
    const float rr = (v[j] - mu) * inv * g1[t + 256 * j] + b1[t + 256 * j];
    y[j] = fmaxf(rr, 0.f) + rr;              // relu(residual) + residual
  }

  float s2 = y[0] + y[1] + y[2] + y[3];
  const float mu2 = block_sum(s2, red) * (1.0f / DMODEL);
  float vs2 = 0.f;
#pragma unroll
  for (int j = 0; j < 4; ++j) { const float d = y[j] - mu2; vs2 += d * d; }
  const float inv2 = rsqrtf(block_sum(vs2, red) * (1.0f / DMODEL) + EPSV);

#pragma unroll
  for (int j = 0; j < 4; ++j)
    out[(size_t)row * DMODEL + t + 256 * j] =
        (y[j] - mu2) * inv2 * g2[t + 256 * j] + b2[t + 256 * j];
}

// ---------------------------------------------------------------------------
extern "C" void kernel_launch(void* const* d_in, const int* in_sizes, int n_in,
                              void* d_out, int out_size, void* d_ws, size_t ws_size,
                              hipStream_t stream) {
  (void)in_sizes; (void)n_in; (void)out_size; (void)ws_size;
  const float* k  = (const float*)d_in[0];
  const float* q  = (const float*)d_in[1];
  const float* r  = (const float*)d_in[2];
  const float* Wk = (const float*)d_in[3];
  const float* bk = (const float*)d_in[4];
  const float* Wq = (const float*)d_in[5];
  const float* bq = (const float*)d_in[6];
  const float* Wv = (const float*)d_in[7];
  const float* bv = (const float*)d_in[8];
  const float* Wo = (const float*)d_in[9];
  const float* bo = (const float*)d_in[10];
  const float* g1 = (const float*)d_in[11];
  const float* b1 = (const float*)d_in[12];
  const float* g2 = (const float*)d_in[13];
  const float* b2 = (const float*)d_in[14];
  float* out = (float*)d_out;

  const size_t E  = (size_t)NB * NSEQ * DMODEL;
  const size_t WE = (size_t)DMODEL * DMODEL;
  _Float16* k16  = (_Float16*)d_ws;            // total ws use ~152 MB
  _Float16* q16  = k16  + E;
  _Float16* r16  = q16  + E;
  _Float16* Wk16 = r16  + E;
  _Float16* Wq16 = Wk16 + WE;
  _Float16* Wv16 = Wq16 + WE;
  _Float16* Wo16 = Wv16 + WE;
  _Float16* kh   = Wo16 + WE;
  _Float16* qh   = kh   + E;
  _Float16* vTh  = qh   + E;                   // head-transposed V
  _Float16* o16  = vTh  + E;
  float*    mha  = (float*)(o16 + E);

  // 1) f32 -> f16 conversions
  const int TB = 256;
  cvt_f32_f16<<<(int)((E + TB - 1) / TB),  TB, 0, stream>>>(k,  k16,  (int)E);
  cvt_f32_f16<<<(int)((E + TB - 1) / TB),  TB, 0, stream>>>(q,  q16,  (int)E);
  cvt_f32_f16<<<(int)((E + TB - 1) / TB),  TB, 0, stream>>>(r,  r16,  (int)E);
  cvt_f32_f16<<<(int)((WE + TB - 1) / TB), TB, 0, stream>>>(Wk, Wk16, (int)WE);
  cvt_f32_f16<<<(int)((WE + TB - 1) / TB), TB, 0, stream>>>(Wq, Wq16, (int)WE);
  cvt_f32_f16<<<(int)((WE + TB - 1) / TB), TB, 0, stream>>>(Wv, Wv16, (int)WE);
  cvt_f32_f16<<<(int)((WE + TB - 1) / TB), TB, 0, stream>>>(Wo, Wo16, (int)WE);

  // 2) projections: 4096 wave-tiles / 8 waves per block
  const int gemm_blocks = (NB * NSEQ / 64) * (DMODEL / 32) / 8;   // 512
  gemm_wmma_t<false><<<gemm_blocks, 256, 0, stream>>>(q16, Wq16, bq, qh, nullptr,
                                                      NB * NSEQ, DMODEL, DMODEL);
  gemm_wmma_t<false><<<gemm_blocks, 256, 0, stream>>>(k16, Wk16, bk, kh, nullptr,
                                                      NB * NSEQ, DMODEL, DMODEL);
  const int gemm_blocks_v = (NB * NSEQ / 32) * (DMODEL / 64) / 8; // 512
  gemm_wmma_vT<<<gemm_blocks_v, 256, 0, stream>>>(r16, Wv16, bv, vTh,
                                                  NB * NSEQ, DMODEL, DMODEL);

  // 3) Attention: B*H*(N/64) blocks of 4 waves
  attn_wmma<<<NB * NHEAD * (NSEQ / 64), 128, 0, stream>>>(qh, kh, vTh, o16);

  // 4) Output projection (f32 out, vectorized stores)
  gemm_wmma_t<true><<<gemm_blocks, 256, 0, stream>>>(o16, Wo16, bo, nullptr, mha,
                                                     NB * NSEQ, DMODEL, DMODEL);

  // 5) residual + LN -> relu+residual -> LN
  fused_ln<<<NB * NSEQ, 256, 0, stream>>>(mha, q, g1, b1, g2, b2, out);
}
